// TreeLSTM_81973745811979
// MI455X (gfx1250) — matrix operand order repaired and checked by reading, not compiled, not run on hardware
//
#include <hip/hip_runtime.h>
#include <hip/hip_bf16.h>

typedef __bf16 bf16_t;
typedef __bf16 v16bf __attribute__((ext_vector_type(16)));
typedef float  v8f   __attribute__((ext_vector_type(8)));
typedef unsigned int u32;
typedef u32 u32x4 __attribute__((ext_vector_type(4)));
typedef u32 u32x8 __attribute__((ext_vector_type(8)));

#define TL_B    128
#define TL_L    512
#define TL_H    128
#define TL_NPT  (2*TL_L-1)      // 1023
#define TL_N    (TL_B*TL_NPT)   // 130944
#define TL_VOCAB 32000
#define TL_OUT  128

union FragU { u32x8 w; u32x4 q[2]; };

__device__ __forceinline__ float sigmoidf_(float x){ return 1.0f/(1.0f+__expf(-x)); }

// Async global->LDS 16B copy (GVS mode: SGPR64 base + per-lane VGPR byte offset).
// The LDS destination operand is the wave-relative LDS offset = low 32 bits of
// the generic shared pointer (flat->LDS mapping truncates to ADDR[31:0]).
__device__ __forceinline__ void async_gather_b128(void* lds_dst, const void* base, u32 byte_off){
    u32 lds_off = (u32)(size_t)lds_dst;
    asm volatile("global_load_async_to_lds_b128 %0, %1, %2"
                 :: "v"(lds_off), "v"(byte_off), "s"(base)
                 : "memory");
}
__device__ __forceinline__ void wait_async0(){
    asm volatile("s_wait_asynccnt 0x0" ::: "memory");
}

// ---------------------------------------------------------------------------
// Prep kernels: fp32 -> bf16 convert, and pack weights into the per-lane
// WMMA B-fragment layout: dst[ct][kt][lane][e], e-th bf16 of a lane is
// src[K = kt*32 + (e>=8 ? 16:0) + (lane>=16 ? 8:0) + (e&7)][N = ct*16 + (lane&15)]  (ISA 7.12.2)
// ---------------------------------------------------------------------------
__global__ void tl_cvt_bf16(const float* __restrict__ src, bf16_t* __restrict__ dst, int n){
    int i = blockIdx.x*blockDim.x + threadIdx.x;
    int stride = gridDim.x*blockDim.x;
    for (; i < n; i += stride) dst[i] = (bf16_t)src[i];
}

__global__ void tl_pack_b(const float* __restrict__ src, bf16_t* __restrict__ dst,
                          int nkt, int ncols, int total){
    int i = blockIdx.x*blockDim.x + threadIdx.x;
    int stride = gridDim.x*blockDim.x;
    for (; i < total; i += stride){
        int el   = i & 15;
        int lane = (i >> 4) & 31;
        int kt   = (i >> 9) % nkt;
        int ct   = (i >> 9) / nkt;
        int K  = kt*32 + ((el >> 3) << 4) + ((lane >> 4) << 3) + (el & 7);
        int Nc = (ct << 4) + (lane & 15);
        dst[i] = (bf16_t)src[(size_t)K*ncols + Nc];
    }
}

// ---------------------------------------------------------------------------
// Leaf kernel: iou = emb[label] @ W_iou + b_iou ; c = sig(i)*tanh(u);
// h = sig(o)*tanh(c).  16 leaves per block, 8 waves; wave w owns hidden tile
// w, computing output col-tiles {w, w+8, w+16} = (i, o, u) for that tile.
// ---------------------------------------------------------------------------
__global__ __launch_bounds__(256) void tl_leaf(
    const int* __restrict__ label,
    const bf16_t* __restrict__ embB,    // [VOCAB][128]
    const bf16_t* __restrict__ WiouF,   // [24][4][32][16]
    const float* __restrict__ b_iou,    // [384]
    bf16_t* __restrict__ hbuf,          // [N][128]
    float*  __restrict__ cbuf)          // [N][128]
{
    __shared__ __attribute__((aligned(32))) bf16_t sA[16*128];
    __shared__ int sNode[16];
    const int tid = threadIdx.x;
    if (tid < 16){
        int gi = blockIdx.x*16 + tid;
        int tree = gi >> 9;
        int li   = gi & 511;
        sNode[tid] = tree*TL_NPT + li;
    }
    {   // async gather of 16 embedding rows (16 x 128 bf16), 16B per thread
        int row = tid >> 4, seg = tid & 15;
        int gi = blockIdx.x*16 + row;
        int nid = (gi >> 9)*TL_NPT + (gi & 511);
        int lab = label[nid];
        async_gather_b128(sA + row*128 + seg*8, embB,
                          (u32)lab*(TL_H*2) + (u32)seg*16);
    }
    wait_async0();
    __syncthreads();

    const int wave = tid >> 5, lane = tid & 31;
    const int arow = lane & 15;
    const int koff = (lane >> 4) << 3;          // 0 or 8
    v16bf af[4];
    #pragma unroll
    for (int kt = 0; kt < 4; ++kt){
        FragU u;
        u.q[0] = *(const u32x4*)(sA + arow*128 + kt*32 + koff);
        u.q[1] = *(const u32x4*)(sA + arow*128 + kt*32 + 16 + koff);
        af[kt] = __builtin_bit_cast(v16bf, u.w);
    }
    v8f acc[3];
    #pragma unroll
    for (int t = 0; t < 3; ++t){
        v8f a = {0.f,0.f,0.f,0.f,0.f,0.f,0.f,0.f};
        int ct = wave + t*8;
        const bf16_t* bp = WiouF + ((size_t)ct*4*32 + lane)*16;
        #pragma unroll
        for (int kt = 0; kt < 4; ++kt){
            v16bf bfr = *(const v16bf*)(bp + (size_t)kt*512);
            a = __builtin_amdgcn_wmma_f32_16x16x32_bf16(
                    false, af[kt], false, bfr, (short)0, a, false, false);
        }
        acc[t] = a;
    }
    const int j = (wave << 4) + (lane & 15);
    const float bi = b_iou[j], bo = b_iou[TL_H + j], bu = b_iou[2*TL_H + j];
    const int mbase = (lane >> 4) << 3;
    #pragma unroll
    for (int r = 0; r < 8; ++r){
        int nid = sNode[mbase + r];
        float iv = sigmoidf_(acc[0][r] + bi);
        float ov = sigmoidf_(acc[1][r] + bo);
        float uv = tanhf(acc[2][r] + bu);
        float cn = iv*uv;
        float hn = ov*tanhf(cn);
        cbuf[(size_t)nid*TL_H + j] = cn;
        hbuf[(size_t)nid*TL_H + j] = (bf16_t)hn;
    }
}

// ---------------------------------------------------------------------------
// Level kernel: h_cat = [h[left]|h[right]] (16x256 LDS tile);
// [fl|fr|i|o|u] = h_cat @ [U_f_w | U_iou] (256x640), gates fused in-register.
// Wave w owns hidden tile w -> col-tiles {w, w+8, w+16, w+24, w+32}.
// ---------------------------------------------------------------------------
__global__ __launch_bounds__(256) void tl_level(
    bf16_t* __restrict__ hbuf, float* __restrict__ cbuf,
    const bf16_t* __restrict__ UF,      // [40][8][32][16]
    const float* __restrict__ U_f_b,    // [256]
    const float* __restrict__ b_iou,    // [384]
    int log2cnt, int noff, int child_off)
{
    __shared__ __attribute__((aligned(32))) bf16_t sA[16*256];
    __shared__ int sIds[48];
    const int tid = threadIdx.x;
    if (tid < 16){
        int gi   = blockIdx.x*16 + tid;
        int tree = gi >> log2cnt;
        int i    = gi - (tree << log2cnt);
        int base = tree*TL_NPT;
        sIds[tid*3+0] = base + noff + i;
        sIds[tid*3+1] = base + child_off + 2*i;
        sIds[tid*3+2] = base + child_off + 2*i + 1;
    }
    #pragma unroll
    for (int it = 0; it < 2; ++it){     // 512 async 16B chunks, 2 per thread
        int ch = tid + it*256;
        int row = ch >> 5, seg = ch & 31;
        int gi   = blockIdx.x*16 + row;
        int tree = gi >> log2cnt;
        int i    = gi - (tree << log2cnt);
        int src  = tree*TL_NPT + child_off + 2*i + (seg >> 4);
        async_gather_b128(sA + row*256 + seg*8, hbuf,
                          (u32)src*(TL_H*2) + (u32)(seg & 15)*16);
    }
    wait_async0();
    __syncthreads();

    const int wave = tid >> 5, lane = tid & 31;
    const int arow = lane & 15;
    const int koff = (lane >> 4) << 3;
    v16bf af[8];
    #pragma unroll
    for (int kt = 0; kt < 8; ++kt){
        FragU u;
        u.q[0] = *(const u32x4*)(sA + arow*256 + kt*32 + koff);
        u.q[1] = *(const u32x4*)(sA + arow*256 + kt*32 + 16 + koff);
        af[kt] = __builtin_bit_cast(v16bf, u.w);
    }
    v8f acc[5];
    #pragma unroll
    for (int t = 0; t < 5; ++t){
        v8f a = {0.f,0.f,0.f,0.f,0.f,0.f,0.f,0.f};
        int ct = wave + t*8;
        const bf16_t* bp = UF + ((size_t)ct*8*32 + lane)*16;
        #pragma unroll
        for (int kt = 0; kt < 8; ++kt){
            v16bf bfr = *(const v16bf*)(bp + (size_t)kt*512);
            a = __builtin_amdgcn_wmma_f32_16x16x32_bf16(
                    false, af[kt], false, bfr, (short)0, a, false, false);
        }
        acc[t] = a;
    }
    const int j = (wave << 4) + (lane & 15);
    const float bfl = U_f_b[j],  bfr2 = U_f_b[TL_H + j];
    const float bi  = b_iou[j],  bo   = b_iou[TL_H + j], bu = b_iou[2*TL_H + j];
    const int mbase = (lane >> 4) << 3;
    #pragma unroll
    for (int r = 0; r < 8; ++r){
        int m = mbase + r;
        int nid = sIds[m*3], lid = sIds[m*3+1], rid = sIds[m*3+2];
        float cl = cbuf[(size_t)lid*TL_H + j];
        float cr = cbuf[(size_t)rid*TL_H + j];
        float fl = sigmoidf_(acc[0][r] + bfl);
        float fr = sigmoidf_(acc[1][r] + bfr2);
        float iv = sigmoidf_(acc[2][r] + bi);
        float ov = sigmoidf_(acc[3][r] + bo);
        float uv = tanhf(acc[4][r] + bu);
        float cn = iv*uv + fl*cl + fr*cr;
        float hn = ov*tanhf(cn);
        cbuf[(size_t)nid*TL_H + j] = cn;
        hbuf[(size_t)nid*TL_H + j] = (bf16_t)hn;
    }
}

// ---------------------------------------------------------------------------
// Root kernel: final = h[roots] @ W_out + b_out; log_softmax over 128 cols.
// ---------------------------------------------------------------------------
__global__ __launch_bounds__(256) void tl_root(
    const bf16_t* __restrict__ hbuf,
    const bf16_t* __restrict__ WoutF,   // [8][4][32][16]
    const float* __restrict__ b_out,    // [128]
    float* __restrict__ out)            // [128][128]
{
    __shared__ __attribute__((aligned(32))) bf16_t sA[16*128];
    __shared__ float sF[16*128];
    __shared__ float sLse[16];
    const int tid = threadIdx.x;
    {
        int row = tid >> 4, seg = tid & 15;
        int tree = blockIdx.x*16 + row;
        int nid  = tree*TL_NPT + (TL_NPT - 1);
        async_gather_b128(sA + row*128 + seg*8, hbuf,
                          (u32)nid*(TL_H*2) + (u32)seg*16);
    }
    wait_async0();
    __syncthreads();
    const int wave = tid >> 5, lane = tid & 31;
    const int arow = lane & 15;
    const int koff = (lane >> 4) << 3;
    v8f acc = {0.f,0.f,0.f,0.f,0.f,0.f,0.f,0.f};
    const bf16_t* bp = WoutF + ((size_t)wave*4*32 + lane)*16;
    #pragma unroll
    for (int kt = 0; kt < 4; ++kt){
        FragU u;
        u.q[0] = *(const u32x4*)(sA + arow*128 + kt*32 + koff);
        u.q[1] = *(const u32x4*)(sA + arow*128 + kt*32 + 16 + koff);
        v16bf af  = __builtin_bit_cast(v16bf, u.w);
        v16bf bfr = *(const v16bf*)(bp + (size_t)kt*512);
        acc = __builtin_amdgcn_wmma_f32_16x16x32_bf16(
                  false, af, false, bfr, (short)0, acc, false, false);
    }
    const int j = (wave << 4) + (lane & 15);
    const float bb = b_out[j];
    const int mbase = (lane >> 4) << 3;
    #pragma unroll
    for (int r = 0; r < 8; ++r)
        sF[(mbase + r)*128 + j] = acc[r] + bb;
    __syncthreads();
    if (tid < 16){
        float mx = -3.4e38f;
        for (int k = 0; k < 128; ++k) mx = fmaxf(mx, sF[tid*128 + k]);
        float s = 0.f;
        for (int k = 0; k < 128; ++k) s += __expf(sF[tid*128 + k] - mx);
        sLse[tid] = mx + __logf(s);
    }
    __syncthreads();
    #pragma unroll
    for (int k = 0; k < 8; ++k){
        int idx = tid + k*256;          // 2048 outputs per block
        int row = idx >> 7, col = idx & 127;
        out[(size_t)(blockIdx.x*16 + row)*TL_OUT + col] = sF[row*128 + col] - sLse[row];
    }
}

// ---------------------------------------------------------------------------
extern "C" void kernel_launch(void* const* d_in, const int* in_sizes, int n_in,
                              void* d_out, int out_size, void* d_ws, size_t ws_size,
                              hipStream_t stream)
{
    (void)in_sizes; (void)n_in; (void)out_size; (void)ws_size;
    const int*   label = (const int*)  d_in[0];
    const float* emb   = (const float*)d_in[1];
    const float* W_iou = (const float*)d_in[2];
    const float* U_iou = (const float*)d_in[3];
    const float* b_iou = (const float*)d_in[4];
    const float* U_f_w = (const float*)d_in[5];
    const float* U_f_b = (const float*)d_in[6];
    const float* W_out = (const float*)d_in[7];
    const float* b_out = (const float*)d_in[8];
    float* out = (float*)d_out;

    char* ws = (char*)d_ws;
    size_t off = 0;
    auto carve = [&](size_t bytes)->char*{
        char* p = ws + off; off = (off + bytes + 255) & ~(size_t)255; return p; };
    bf16_t* hbuf  = (bf16_t*)carve((size_t)TL_N*TL_H*2);       // 33.5 MB
    float*  cbuf  = (float*) carve((size_t)TL_N*TL_H*4);       // 67.0 MB
    bf16_t* embB  = (bf16_t*)carve((size_t)TL_VOCAB*TL_H*2);   //  8.2 MB
    bf16_t* UF    = (bf16_t*)carve((size_t)40*8*512*2);        // 320 KB
    bf16_t* WiouF = (bf16_t*)carve((size_t)24*4*512*2);        //  96 KB
    bf16_t* WoutF = (bf16_t*)carve((size_t)8*4*512*2);         //  32 KB

    tl_cvt_bf16<<<1024, 256, 0, stream>>>(emb, embB, TL_VOCAB*TL_H);
    tl_pack_b<<<128, 256, 0, stream>>>(U_f_w, UF,            8, 256, 16*8*512);
    tl_pack_b<<<192, 256, 0, stream>>>(U_iou, UF + 16*8*512, 8, 384, 24*8*512);
    tl_pack_b<<< 96, 256, 0, stream>>>(W_iou, WiouF,         4, 384, 24*4*512);
    tl_pack_b<<< 32, 256, 0, stream>>>(W_out, WoutF,         4, 128,  8*4*512);

    tl_leaf<<<(TL_B*TL_L)/16, 256, 0, stream>>>(label, embB, WiouF, b_iou, hbuf, cbuf);

    for (int lvl = 1; lvl <= 9; ++lvl){
        int cnt      = TL_L >> lvl;                 // 256 .. 1
        int log2cnt  = 9 - lvl;
        int noff     = 2*TL_L - ((2*TL_L) >> lvl);  // 512, 768, ..., 1022
        int child    = 2*TL_L - ((4*TL_L) >> lvl);  // 0, 512, ..., 1020
        int blocks   = (TL_B*cnt)/16;               // always multiple of 16 nodes
        tl_level<<<blocks, 256, 0, stream>>>(hbuf, cbuf, UF, U_f_b, b_iou,
                                             log2cnt, noff, child);
    }
    tl_root<<<TL_B/16, 256, 0, stream>>>(hbuf, WoutF, b_out, out);
}
